// CSAttention_70463233458347
// MI455X (gfx1250) — matrix operand order repaired and checked by reading, not compiled, not run on hardware
//
#include <hip/hip_runtime.h>
#include <math.h>

// ---------------- problem constants (from reference setup_inputs) ----------
#define BATCH 4
#define CIN   64
#define HH    192
#define WW    192
#define HWD   (HH*WW)      // 36864 pixels per image
#define HDIM  128
#define NWX   24           // windows per side
#define NWIN  576          // 24*24 windows
#define HEADS 4

typedef _Float16 f16;
typedef _Float16 v16h __attribute__((ext_vector_type(16)));
typedef float    v8f  __attribute__((ext_vector_type(8)));

// ---------------- WMMA helpers (wave32, 16x16x32 f16 -> f32) ---------------
__device__ __forceinline__ v8f wmma16(v16h a, v16h b, v8f c) {
  return __builtin_amdgcn_wmma_f32_16x16x32_f16(false, a, false, b, (short)0, c,
                                                false, false);
}

// A fragment: 16x32 f16, lds row-major [row][k], contiguous k order (same as B):
// lanes 0-15: row, K = 0..15+k0 ; lanes 16-31: K = 16..31+k0
template<int KMAX>
__device__ __forceinline__ v16h load_a_frag(const f16* lds, int row0, int stride, int k0) {
  int lane = threadIdx.x & 31;
  const f16* p = lds + (row0 + (lane & 15)) * stride;
  int kb = k0 + ((lane >> 4) << 4);
  v16h a;
#pragma unroll
  for (int j = 0; j < 16; ++j) {
    int k = kb + j;
    a[j] = (k < KMAX) ? p[k] : (f16)0.f;
  }
  return a;
}

// B fragment: 32x16 f16. Stored transposed in LDS as [col][k] (contiguous k).
// lanes 0-15: col, K = 0..15+k0 ; lanes 16-31: col, K = 16..31+k0
template<int KMAX>
__device__ __forceinline__ v16h load_b_frag(const f16* lds, int col0, int stride, int k0) {
  int lane = threadIdx.x & 31;
  const f16* p = lds + (col0 + (lane & 15)) * stride;
  int kb = k0 + ((lane >> 4) << 4);
  v16h b;
#pragma unroll
  for (int j = 0; j < 16; ++j) {
    int k = kb + j;
    b[j] = (k < KMAX) ? p[k] : (f16)0.f;
  }
  return b;
}

// rolled coordinate: window frame (after roll by -SS) -> planar pixel index.
// The inverse un-roll (+SS) uses the identical mapping.
__device__ __forceinline__ int rollpix(int wy, int wx, int ky, int kx) {
  int y = (wy * 8 + ky + 4) % HH;
  int x = (wx * 8 + kx + 4) % WW;
  return y * WW + x;
}

// ===========================================================================
// K0: fused QUZ + KV projection.  C[640 x 147456] = W[640x64] * X[64 x Npix]
// f16 WMMA, f32 accum, f16 outputs (q,u,z,k,v planar).
// ===========================================================================
__global__ __launch_bounds__(256) void k_proj(const float* __restrict__ x,
                                              const float* __restrict__ wquz,
                                              const float* __restrict__ wkv,
                                              f16* qh, f16* uh, f16* zh,
                                              f16* kh, f16* vh) {
  __shared__ f16 Xl[128 * 72];   // [pixel][ch]   (B-frag layout)
  __shared__ f16 Wl[128 * 72];   // [row][ch]     (A-frag layout), 128-row chunk
  int bid = blockIdx.x;               // 1152 = 4 * 288
  int b = bid / 288;
  int hw0 = (bid % 288) * 128;
  int t = threadIdx.x;
  int wave = t >> 5, lane = t & 31;

  for (int id = t; id < 128 * 64; id += 256) {
    int ch = id >> 7, px = id & 127;
    Xl[px * 72 + ch] = (f16)x[(b * 64 + ch) * HWD + hw0 + px];
  }

  for (int ci = 0; ci < 5; ++ci) {      // 5 chunks of 128 output channels
    __syncthreads();
    for (int id = t; id < 128 * 64; id += 256) {
      int r = id >> 6, k = id & 63;
      int row = ci * 128 + r;
      float wv = (row < 384) ? wquz[row * 64 + k] : wkv[(row - 384) * 64 + k];
      Wl[r * 72 + k] = (f16)wv;
    }
    __syncthreads();
    f16* dst = (ci == 0) ? qh : (ci == 1) ? uh : (ci == 2) ? zh : (ci == 3) ? kh : vh;
    v16h a0 = load_a_frag<64>(Wl, wave * 16, 72, 0);
    v16h a1 = load_a_frag<64>(Wl, wave * 16, 72, 32);
    int cstore = lane & 15;
    int rbase  = (lane >> 4) << 3;
    for (int nt = 0; nt < 8; ++nt) {
      v16h b0 = load_b_frag<64>(Xl, nt * 16, 72, 0);
      v16h b1 = load_b_frag<64>(Xl, nt * 16, 72, 32);
      v8f acc = {};
      acc = wmma16(a0, b0, acc);
      acc = wmma16(a1, b1, acc);
      int px = nt * 16 + cstore;
#pragma unroll
      for (int r = 0; r < 8; ++r) {
        int ch = wave * 16 + rbase + r;
        dst[(b * HDIM + ch) * HWD + hw0 + px] = (f16)acc[r];
      }
    }
  }
}

// ===========================================================================
// K1: per-window attn_b raw (WMMA 64x64, K=8) + attn_l raw (8x8, K=64)
// One block per (b, head, window); includes L2 normalization.
// ===========================================================================
__global__ __launch_bounds__(128) void k_attnb(const f16* __restrict__ qh,
                                               const f16* __restrict__ kh,
                                               const float* __restrict__ temp,
                                               f16* attnb, float* attnl_raw) {
  int bid = blockIdx.x;                // (b*4+h)*576 + n
  int n = bid % NWIN; int bh = bid / NWIN; int h = bh & 3, b = bh >> 2;
  int wy = n / NWX, wx = n % NWX;
  __shared__ float qbr[512], kbr[512], qlr[512], klr[512]; // [pos][c], c=0..7
  __shared__ f16 qbn[64 * 12], kbn[64 * 12];
  __shared__ float nlq[8], nlk[8];
  int t = threadIdx.x;

  for (int id = t; id < 512; id += 128) {
    int i = id >> 3, c = id & 7;
    int p = rollpix(wy, wx, i >> 3, i & 7);
    int base = (b * HDIM + h * 32) * HWD;
    qbr[id] = (float)qh[base + c * HWD + p];
    kbr[id] = (float)kh[base + c * HWD + p];
    qlr[id] = (float)qh[base + (8 + c) * HWD + p];
    klr[id] = (float)kh[base + (8 + c) * HWD + p];
  }
  __syncthreads();
  if (t < 64) {                         // normalize b-branch per position
    float sq = 0.f, sk = 0.f;
    for (int c = 0; c < 8; ++c) { float a = qbr[t*8+c]; sq += a*a;
                                  float e = kbr[t*8+c]; sk += e*e; }
    float rq = 1.f / fmaxf(sqrtf(sq), 1e-12f);
    float rk = 1.f / fmaxf(sqrtf(sk), 1e-12f);
    for (int c = 0; c < 8; ++c) { qbn[t*12+c] = (f16)(qbr[t*8+c]*rq);
                                  kbn[t*12+c] = (f16)(kbr[t*8+c]*rk); }
    for (int c = 8; c < 12; ++c) { qbn[t*12+c] = (f16)0.f; kbn[t*12+c] = (f16)0.f; }
  } else if (t < 72) {                  // l-branch norms over positions
    int c = t - 64; float sq = 0.f, sk = 0.f;
    for (int i = 0; i < 64; ++i) { float a = qlr[i*8+c]; sq += a*a;
                                   float e = klr[i*8+c]; sk += e*e; }
    nlq[c] = fmaxf(sqrtf(sq), 1e-12f);
    nlk[c] = fmaxf(sqrtf(sk), 1e-12f);
  }
  __syncthreads();

  if (t < 64) {                         // attn_l raw (c,d), K = 64 positions
    int c = t >> 3, d = t & 7;
    float s = 0.f;
    for (int i = 0; i < 64; ++i) s += qlr[i*8+c] * klr[i*8+d];
    attnl_raw[(size_t)bid * 64 + t] = s / (nlq[c] * nlk[d]) * temp[1*HEADS + h];
  }

  float t0 = temp[0*HEADS + h];
  int wave = t >> 5, lane = t & 31;     // 4 waves, wave w -> i-tile w
  v16h a = load_a_frag<8>(qbn, wave * 16, 12, 0);
  for (int jt = 0; jt < 4; ++jt) {
    v16h bb = load_b_frag<8>(kbn, jt * 16, 12, 0);
    v8f acc = {};
    acc = wmma16(a, bb, acc);
    int col = jt * 16 + (lane & 15);
    int rb  = (lane >> 4) << 3;
#pragma unroll
    for (int r = 0; r < 8; ++r) {
      int row = wave * 16 + rb + r;
      attnb[(size_t)bid * 4096 + row * 64 + col] = (f16)(acc[r] * t0);
    }
  }
}

// ===========================================================================
// K2: attn_l talking conv (3x3 over window grid, heads mixed) + softmax(d)
// ===========================================================================
__global__ __launch_bounds__(256) void k_attnl_talk(const float* __restrict__ attnl_raw,
                                                    const float* __restrict__ wl,
                                                    float* attnl_fin) {
  int bid = blockIdx.x;                 // b*576+n
  int n = bid % NWIN, b = bid / NWIN;
  int wy = n / NWX, wx = n % NWX;
  int t = threadIdx.x;                  // th*64 + c*8 + d
  int th = t >> 6, c = (t >> 3) & 7, d = t & 7;
  float s = 0.f;
  for (int h = 0; h < 4; ++h)
    for (int dy = 0; dy < 3; ++dy) {
      int yy = wy + dy - 1; if (yy < 0 || yy >= NWX) continue;
      for (int dx = 0; dx < 3; ++dx) {
        int xx = wx + dx - 1; if (xx < 0 || xx >= NWX) continue;
        s += wl[((th*4 + h)*3 + dy)*3 + dx] *
             attnl_raw[((size_t)(b*4 + h) * NWIN + yy*NWX + xx) * 64 + c*8 + d];
      }
    }
  __shared__ float vals[256];
  vals[t] = s;
  __syncthreads();
  int rb = t & ~7;
  float m = -1e30f;
  for (int j = 0; j < 8; ++j) m = fmaxf(m, vals[rb + j]);
  float sum = 0.f;
  for (int j = 0; j < 8; ++j) sum += expf(vals[rb + j] - m);
  attnl_fin[((size_t)(b*4 + th) * NWIN + n) * 64 + c*8 + d] = expf(s - m) / sum;
}

// ===========================================================================
// K3: attn_g raw: per (b,h,i) reduce over all 576 windows (norm folded after)
// ===========================================================================
__global__ __launch_bounds__(64) void k_attng_raw(const f16* __restrict__ qh,
                                                  const f16* __restrict__ kh,
                                                  const float* __restrict__ temp,
                                                  float* attng_raw) {
  int bid = blockIdx.x;                 // (b*4+h)*64 + i
  int i = bid & 63; int bh = bid >> 6; int h = bh & 3, b = bh >> 2;
  int ky = i >> 3, kx = i & 7;
  __shared__ f16 qg[NWIN * 8], kg[NWIN * 8];
  int t = threadIdx.x;
  for (int id = t; id < NWIN * 8; id += 64) {
    int n = id >> 3, c = id & 7;
    int p = rollpix(n / NWX, n % NWX, ky, kx);
    int base = (b * HDIM + h * 32 + 16) * HWD;
    qg[id] = qh[base + c * HWD + p];
    kg[id] = kh[base + c * HWD + p];
  }
  __syncthreads();
  int c = t >> 3, d = t & 7;
  float s = 0.f, sq = 0.f, sk = 0.f;
  for (int n = 0; n < NWIN; ++n) {
    float a = (float)qg[n*8 + c], e = (float)kg[n*8 + d];
    s += a * e; sq += a * a; sk += e * e;
  }
  float nq = fmaxf(sqrtf(sq), 1e-12f), nk = fmaxf(sqrtf(sk), 1e-12f);
  attng_raw[(size_t)bid * 64 + t] = s / (nq * nk) * temp[2*HEADS + h];
}

// K3b: g-branch talking (64x64 mix over (h,k)->(t,l)) + softmax(d)
__global__ __launch_bounds__(64) void k_attng_talk(const float* __restrict__ attng_raw,
                                                   const float* __restrict__ g,
                                                   float* attng_fin) {
  int bid = blockIdx.x;                 // (b*4+t)*64 + l
  int l = bid & 63; int bt = bid >> 6; int th = bt & 3, b = bt >> 2;
  int t = threadIdx.x;                  // c*8+d
  float s = 0.f;
  for (int h = 0; h < 4; ++h)
    for (int k = 0; k < 64; ++k)
      s += g[((h*64 + k)*64 + l)*4 + th] *
           attng_raw[((size_t)(b*4 + h) * 64 + k) * 64 + t];
  __shared__ float vals[64];
  vals[t] = s;
  __syncthreads();
  int rb = t & ~7;
  float m = -1e30f;
  for (int j = 0; j < 8; ++j) m = fmaxf(m, vals[rb + j]);
  float sum = 0.f;
  for (int j = 0; j < 8; ++j) sum += expf(vals[rb + j] - m);
  attng_fin[(size_t)bid * 64 + t] = expf(s - m) / sum;
}

// ===========================================================================
// K4: attn_r raw: per (b,h,c) a 64x64 GEMM with K = 576 windows -> WMMA,
// chunked K=32 through LDS; norms via shared-atomic accumulation.
// ===========================================================================
__global__ __launch_bounds__(128) void k_attnr_raw(const f16* __restrict__ qh,
                                                   const f16* __restrict__ kh,
                                                   const float* __restrict__ temp,
                                                   float* attnr_raw) {
  int bid = blockIdx.x;                 // (b*4+h)*8 + c
  int c = bid & 7; int bh = bid >> 3; int h = bh & 3, b = bh >> 2;
  __shared__ f16 Qs[64 * 36], Ks[64 * 36];
  __shared__ float nq2[64], nk2[64];
  int t = threadIdx.x;
  if (t < 64) { nq2[t] = 0.f; nk2[t] = 0.f; }
  int wave = t >> 5, lane = t & 31;
  int plane = (b * HDIM + h * 32 + 24 + c) * HWD;
  v8f acc[4] = {};
  for (int cn = 0; cn < 18; ++cn) {     // 18 * 32 = 576
    __syncthreads();
    for (int id = t; id < 64 * 32; id += 128) {
      int i = id >> 5, nn = id & 31;
      int n = cn * 32 + nn;
      int p = rollpix(n / NWX, n % NWX, i >> 3, i & 7);
      float qv = (float)qh[plane + p];
      float kv = (float)kh[plane + p];
      Qs[i * 36 + nn] = (f16)qv;
      Ks[i * 36 + nn] = (f16)kv;
      atomicAdd(&nq2[i], qv * qv);
      atomicAdd(&nk2[i], kv * kv);
    }
    __syncthreads();
    v16h a = load_a_frag<32>(Qs, wave * 16, 36, 0);
#pragma unroll
    for (int jt = 0; jt < 4; ++jt) {
      v16h bb = load_b_frag<32>(Ks, jt * 16, 36, 0);
      acc[jt] = wmma16(a, bb, acc[jt]);
    }
  }
  __syncthreads();
  float t3 = temp[3*HEADS + h];
  int cb = lane & 15, rb = (lane >> 4) << 3;
#pragma unroll
  for (int jt = 0; jt < 4; ++jt) {
    int col = jt * 16 + cb;
    float rk = 1.f / fmaxf(sqrtf(nk2[col]), 1e-12f);
#pragma unroll
    for (int r = 0; r < 8; ++r) {
      int row = wave * 16 + rb + r;
      float rq = 1.f / fmaxf(sqrtf(nq2[row]), 1e-12f);
      attnr_raw[(size_t)bid * 4096 + row * 64 + col] = acc[jt][r] * rq * rk * t3;
    }
  }
}

// K4b: r-branch talking ((h,c)->(t,d) mix) + softmax over j
__global__ __launch_bounds__(64) void k_attnr_talk(const float* __restrict__ attnr_raw,
                                                   const float* __restrict__ rt,
                                                   float* attnr_fin) {
  int bid = blockIdx.x;                 // ((b*4+t)*8+d)*64 + i
  int i = bid & 63; int r1 = bid >> 6; int d = r1 & 7;
  int bt = r1 >> 3; int th = bt & 3, b = bt >> 2;
  int j = threadIdx.x;
  float s = 0.f;
  for (int h = 0; h < 4; ++h)
    for (int c = 0; c < 8; ++c)
      s += rt[((h*8 + c)*8 + d)*4 + th] *
           attnr_raw[(size_t)((b*4 + h)*8 + c) * 4096 + i * 64 + j];
  __shared__ float vals[64];
  vals[j] = s;
  __syncthreads();
  float m = -1e30f;
  for (int q = 0; q < 64; ++q) m = fmaxf(m, vals[q]);
  float sum = 0.f;
  for (int q = 0; q < 64; ++q) sum += expf(vals[q] - m);
  attnr_fin[(size_t)bid * 64 + j] = expf(s - m) / sum;
}

// ===========================================================================
// K5: fused window output: b-branch talking conv (9 neighbors x 4 heads via
// LDS) + shift mask + softmax + WMMA out_b, plus VALU out_l/out_g/out_r.
// One block per (b, window, 16-row slice). Writes un-rolled planar output.
// ===========================================================================
__global__ __launch_bounds__(256) void k_winout(const f16* __restrict__ attnb,
                                                const f16* __restrict__ vh,
                                                const float* __restrict__ attnl_fin,
                                                const float* __restrict__ attng_fin,
                                                const float* __restrict__ attnr_fin,
                                                const float* __restrict__ wb,
                                                float* attn_out) {
  int bid = blockIdx.x;                 // ((b*576+n)*4 + it)
  int it = bid & 3; int bn = bid >> 2;
  int n = bn % NWIN, b = bn / NWIN;
  int wy = n / NWX, wx = n % NWX;
  const int i0 = it * 16;

  __shared__ f16   Nb[9 * 1024];        // 9 neighbor slices [16 x 64] (one head)
  __shared__ float accT[4][1024];       // conv accum, 4 output heads
  __shared__ f16   Asm[16 * 72];        // softmaxed attn slice (A-frag)
  __shared__ f16   vT[32 * 72];         // v window transposed [ch][pos]
  __shared__ float segred[16][16];
  int t = threadIdx.x;

  for (int id = t; id < 4 * 1024; id += 256) ((float*)accT)[id] = 0.f;

  for (int h = 0; h < 4; ++h) {         // accumulate talking conv per in-head
    __syncthreads();
    for (int id = t; id < 9 * 1024; id += 256) {
      int nb = id >> 10; int rem = id & 1023;
      int i = rem >> 6, j = rem & 63;
      int yy = wy + nb / 3 - 1, xx = wx + nb % 3 - 1;
      f16 v = (f16)0.f;
      if (yy >= 0 && yy < NWX && xx >= 0 && xx < NWX)
        v = attnb[((size_t)(b*4 + h) * NWIN + yy*NWX + xx) * 4096 +
                  (i0 + i) * 64 + j];
      Nb[id] = v;
    }
    __syncthreads();
    for (int id = t; id < 4 * 1024; id += 256) {
      int th = id >> 10; int rem = id & 1023;
      float s = 0.f;
#pragma unroll
      for (int nb = 0; nb < 9; ++nb)
        s += wb[((th*4 + h)*3 + nb/3)*3 + nb%3] * (float)Nb[nb * 1024 + rem];
      accT[th][rem] += s;
    }
  }
  __syncthreads();

  for (int th = 0; th < 4; ++th) {      // per output head
    __syncthreads();
    for (int id = t; id < 32 * 64; id += 256) {   // stage v (all 4 branches)
      int ch = id >> 6, pos = id & 63;
      int p = rollpix(wy, wx, pos >> 3, pos & 7);
      vT[ch * 72 + pos] = vh[(b * HDIM + th * 32 + ch) * HWD + p];
    }
    __syncthreads();

    // mask + softmax over j (rows i0..i0+15); thread = row(4b) x seg(4b)
    int r = t >> 4, seg = t & 15;
    int gi = i0 + r;
    int rh_i = (wy < 23) ? 0 : (((gi >> 3) < 4) ? 1 : 2);
    int rw_i = (wx < 23) ? 0 : (((gi & 7) < 4) ? 1 : 2);
    float lv[4]; float lm = -1e30f;
#pragma unroll
    for (int q = 0; q < 4; ++q) {
      int j = seg * 4 + q;
      int rh_j = (wy < 23) ? 0 : (((j >> 3) < 4) ? 1 : 2);
      int rw_j = (wx < 23) ? 0 : (((j & 7) < 4) ? 1 : 2);
      float mask = ((rh_i*3 + rw_i) != (rh_j*3 + rw_j)) ? -100.f : 0.f;
      lv[q] = accT[th][r * 64 + j] + mask;
      lm = fmaxf(lm, lv[q]);
    }
    segred[r][seg] = lm; __syncthreads();
    float m = -1e30f;
    for (int q = 0; q < 16; ++q) m = fmaxf(m, segred[r][q]);
    float ls = 0.f;
#pragma unroll
    for (int q = 0; q < 4; ++q) { lv[q] = expf(lv[q] - m); ls += lv[q]; }
    __syncthreads();
    segred[r][seg] = ls; __syncthreads();
    float sum = 0.f;
    for (int q = 0; q < 16; ++q) sum += segred[r][q];
    float inv = 1.f / sum;
#pragma unroll
    for (int q = 0; q < 4; ++q) Asm[r * 72 + seg * 4 + q] = (f16)(lv[q] * inv);
    __syncthreads();

    if ((t >> 5) == 0) {                // wave 0: out_b = attn(16x64) x vb(64x8)
      v16h a0 = load_a_frag<64>(Asm, 0, 72, 0);
      v16h a1 = load_a_frag<64>(Asm, 0, 72, 32);
      v16h b0 = load_b_frag<64>(vT, 0, 72, 0);
      v16h b1 = load_b_frag<64>(vT, 0, 72, 32);
      v8f acc = {};
      acc = wmma16(a0, b0, acc);
      acc = wmma16(a1, b1, acc);
      int lane = t & 31; int cc = lane & 15; int rb2 = (lane >> 4) << 3;
      if (cc < 8) {
#pragma unroll
        for (int r2 = 0; r2 < 8; ++r2) {
          int i = i0 + rb2 + r2;
          int p = rollpix(wy, wx, i >> 3, i & 7);
          attn_out[(b * HDIM + th * 8 + cc) * HWD + p] = acc[r2];
        }
      }
    } else if (t < 160) {               // waves 1-4: out_r (16 i x 8 c)
      int e = t - 32; int li = e >> 3, c = e & 7;
      int i = i0 + li;
      float s = 0.f;
      const float* Ar = attnr_fin + (size_t)((b*4 + th)*8 + c) * 4096 + i * 64;
      for (int j = 0; j < 64; ++j) s += Ar[j] * (float)vT[(24 + c) * 72 + j];
      int p = rollpix(wy, wx, i >> 3, i & 7);
      attn_out[(b * HDIM + 96 + th * 8 + c) * HWD + p] = s;
    } else {                            // waves 5-7: out_l + out_g (2 x 128)
      for (int id = t - 160; id < 256; id += 96) {
        int e = id & 127; int li = e >> 3, c = e & 7;
        int i = i0 + li;
        float s = 0.f;
        if (id < 128) {
          const float* Al = attnl_fin + ((size_t)(b*4 + th) * NWIN + n) * 64 + c * 8;
          for (int d = 0; d < 8; ++d) s += Al[d] * (float)vT[(8 + d) * 72 + i];
          int p = rollpix(wy, wx, i >> 3, i & 7);
          attn_out[(b * HDIM + 32 + th * 8 + c) * HWD + p] = s;
        } else {
          const float* Ag = attng_fin + ((size_t)(b*4 + th) * 64 + i) * 64 + c * 8;
          for (int d = 0; d < 8; ++d) s += Ag[d] * (float)vT[(16 + d) * 72 + i];
          int p = rollpix(wy, wx, i >> 3, i & 7);
          attn_out[(b * HDIM + 64 + th * 8 + c) * HWD + p] = s;
        }
      }
    }
  }
}

// ===========================================================================
// K6/K7: channel attention (global mean + 1x1)
// ===========================================================================
__global__ __launch_bounds__(256) void k_mean(const float* __restrict__ attn_out,
                                              float* meanv) {
  int bc = blockIdx.x;                  // b*128+ch
  const float* p = attn_out + (size_t)bc * HWD;
  float s = 0.f;
  for (int i = threadIdx.x; i < HWD; i += 256) {
    if (i + 4096 < HWD) __builtin_prefetch(p + i + 4096, 0, 1);
    s += p[i];
  }
  __shared__ float red[256];
  red[threadIdx.x] = s;
  __syncthreads();
  for (int off = 128; off > 0; off >>= 1) {
    if (threadIdx.x < off) red[threadIdx.x] += red[threadIdx.x + off];
    __syncthreads();
  }
  if (threadIdx.x == 0) meanv[bc] = red[0] / (float)HWD;
}

__global__ __launch_bounds__(512) void k_sca(const float* __restrict__ meanv,
                                             const float* __restrict__ sw,
                                             const float* __restrict__ sb,
                                             float* svec) {
  int t = threadIdx.x; int b = t >> 7, o = t & 127;
  float s = sb[o];
  for (int c = 0; c < 128; ++c) s += sw[o * 128 + c] * meanv[b * 128 + c];
  svec[t] = s;
}

// ===========================================================================
// K8: epilogue: project_w @ (out*s)  [+dwconv(u)] * z  -> project_out @ (.)
// Two WMMA GEMMs back-to-back through LDS, 128-pixel tiles.
// ===========================================================================
__global__ __launch_bounds__(256) void k_final(const float* __restrict__ attn_out,
                                               const float* __restrict__ svec,
                                               const float* __restrict__ pw,
                                               const float* __restrict__ pow_,
                                               const float* __restrict__ dww,
                                               const f16* __restrict__ uh,
                                               const f16* __restrict__ zh,
                                               float* outp) {
  __shared__ f16 T[128 * 132];          // T1 [px][c] then T2 [px][o]
  __shared__ f16 Wp[64 * 132];          // weight chunk [row][k]
  int bid = blockIdx.x;                 // 1152
  int b = bid / 288; int hw0 = (bid % 288) * 128;
  int t = threadIdx.x; int wave = t >> 5, lane = t & 31;
  int ntb = (wave & 1) * 4;
  int rb2 = (lane >> 4) << 3, cb2 = lane & 15;

  for (int id = t; id < 128 * 128; id += 256) {
    int px = id >> 7, c = id & 127;
    T[px * 132 + c] = (f16)(attn_out[(size_t)(b * HDIM + c) * HWD + hw0 + px]
                            * svec[b * HDIM + c]);
  }

  v8f acc[8];
  for (int mc = 0; mc < 2; ++mc) {      // GEMM1, M = 128 in two 64-row chunks
    __syncthreads();
    for (int id = t; id < 64 * 128; id += 256) {
      int r = id >> 7, c = id & 127;
      Wp[r * 132 + c] = (f16)pw[(mc * 64 + r) * 128 + c];
    }
    __syncthreads();
    int mt = wave >> 1;
    v16h a[4];
#pragma unroll
    for (int kk = 0; kk < 4; ++kk) a[kk] = load_a_frag<128>(Wp, mt * 16, 132, kk * 32);
#pragma unroll
    for (int q = 0; q < 4; ++q) {
      int nt = ntb + q;
      v8f cc = {};
#pragma unroll
      for (int kk = 0; kk < 4; ++kk) {
        v16h bb = load_b_frag<128>(T, nt * 16, 132, kk * 32);
        cc = wmma16(a[kk], bb, cc);
      }
      acc[mc * 4 + q] = cc;
    }
  }
  __syncthreads();

  // T2[px][o] = (gemm1 + depthwise(u)) * z
  int mt = wave >> 1;
#pragma unroll
  for (int mc = 0; mc < 2; ++mc) {
#pragma unroll
    for (int q = 0; q < 4; ++q) {
      int px = (ntb + q) * 16 + cb2;
      int pix = hw0 + px; int y = pix / WW, x = pix % WW;
#pragma unroll
      for (int r = 0; r < 8; ++r) {
        int o = mc * 64 + mt * 16 + rb2 + r;
        float dwv = 0.f;
        for (int dy = 0; dy < 3; ++dy) {
          int yy = y + dy - 1; if (yy < 0 || yy >= HH) continue;
          for (int dx = 0; dx < 3; ++dx) {
            int xx = x + dx - 1; if (xx < 0 || xx >= WW) continue;
            dwv += dww[o * 9 + dy * 3 + dx] *
                   (float)uh[(size_t)(b * HDIM + o) * HWD + yy * WW + xx];
          }
        }
        float zv = (float)zh[(size_t)(b * HDIM + o) * HWD + pix];
        T[px * 132 + o] = (f16)((acc[mc * 4 + q][r] + dwv) * zv);
      }
    }
  }
  __syncthreads();
  for (int id = t; id < 64 * 128; id += 256) {
    int r = id >> 7, c = id & 127;
    Wp[r * 132 + c] = (f16)pow_[r * 128 + c];
  }
  __syncthreads();

  int mt2 = wave >> 1;                  // GEMM2: M = 64, K = 128
  v16h a2[4];
#pragma unroll
  for (int kk = 0; kk < 4; ++kk) a2[kk] = load_a_frag<128>(Wp, mt2 * 16, 132, kk * 32);
#pragma unroll
  for (int q = 0; q < 4; ++q) {
    int nt = ntb + q;
    v8f cc = {};
#pragma unroll
    for (int kk = 0; kk < 4; ++kk) {
      v16h bb = load_b_frag<128>(T, nt * 16, 132, kk * 32);
      cc = wmma16(a2[kk], bb, cc);
    }
    int px = nt * 16 + cb2;
#pragma unroll
    for (int r = 0; r < 8; ++r) {
      int o = mt2 * 16 + rb2 + r;
      outp[(size_t)(b * 64 + o) * HWD + hw0 + px] = cc[r];
    }
  }
}

// ===========================================================================
extern "C" void kernel_launch(void* const* d_in, const int* in_sizes, int n_in,
                              void* d_out, int out_size, void* d_ws, size_t ws_size,
                              hipStream_t stream) {
  (void)in_sizes; (void)n_in; (void)out_size; (void)ws_size;
  const float* x       = (const float*)d_in[0];
  const float* w_quz   = (const float*)d_in[1];
  const float* w_kv    = (const float*)d_in[2];
  const float* temp    = (const float*)d_in[3];
  const float* r_talk  = (const float*)d_in[4];
  const float* g_talk  = (const float*)d_in[5];
  const float* b_talk  = (const float*)d_in[6];
  const float* l_talk  = (const float*)d_in[7];
  const float* dw_w    = (const float*)d_in[8];
  const float* proj_w  = (const float*)d_in[9];
  const float* pout_w  = (const float*)d_in[10];
  const float* sca_w   = (const float*)d_in[11];
  const float* sca_b   = (const float*)d_in[12];

  // workspace carve-out (~340 MB total)
  char* w = (char*)d_ws;
  auto alloc = [&](size_t bytes) {
    char* p = w; w += (bytes + 255) & ~(size_t)255; return p;
  };
  const size_t PLANAR = (size_t)BATCH * HDIM * HWD;   // 18,874,368 elems
  f16*   qh        = (f16*)alloc(PLANAR * 2);
  f16*   uh        = (f16*)alloc(PLANAR * 2);
  f16*   zh        = (f16*)alloc(PLANAR * 2);
  f16*   kh        = (f16*)alloc(PLANAR * 2);
  f16*   vh        = (f16*)alloc(PLANAR * 2);
  f16*   attnb     = (f16*)alloc((size_t)BATCH * HEADS * NWIN * 4096 * 2);
  float* attnl_raw = (float*)alloc((size_t)BATCH * HEADS * NWIN * 64 * 4);
  float* attnl_fin = (float*)alloc((size_t)BATCH * HEADS * NWIN * 64 * 4);
  float* attng_raw = (float*)alloc((size_t)BATCH * HEADS * 64 * 64 * 4);
  float* attng_fin = (float*)alloc((size_t)BATCH * HEADS * 64 * 64 * 4);
  float* attnr_raw = (float*)alloc((size_t)BATCH * HEADS * 8 * 4096 * 4);
  float* attnr_fin = (float*)alloc((size_t)BATCH * HEADS * 8 * 4096 * 4);
  float* attn_out  = (float*)alloc(PLANAR * 4);
  float* meanv     = (float*)alloc(BATCH * HDIM * 4);
  float* svec      = (float*)alloc(BATCH * HDIM * 4);

  k_proj      <<<1152, 256, 0, stream>>>(x, w_quz, w_kv, qh, uh, zh, kh, vh);
  k_attnb     <<<BATCH * HEADS * NWIN, 128, 0, stream>>>(qh, kh, temp, attnb, attnl_raw);
  k_attnl_talk<<<BATCH * NWIN, 256, 0, stream>>>(attnl_raw, l_talk, attnl_fin);
  k_attng_raw <<<BATCH * HEADS * 64, 64, 0, stream>>>(qh, kh, temp, attng_raw);
  k_attng_talk<<<BATCH * HEADS * 64, 64, 0, stream>>>(attng_raw, g_talk, attng_fin);
  k_attnr_raw <<<BATCH * HEADS * 8, 128, 0, stream>>>(qh, kh, temp, attnr_raw);
  k_attnr_talk<<<BATCH * HEADS * 8 * 64, 64, 0, stream>>>(attnr_raw, r_talk, attnr_fin);
  k_winout    <<<BATCH * NWIN * 4, 256, 0, stream>>>(attnb, vh, attnl_fin, attng_fin,
                                                     attnr_fin, b_talk, attn_out);
  k_mean      <<<BATCH * HDIM, 256, 0, stream>>>(attn_out, meanv);
  k_sca       <<<1, 512, 0, stream>>>(meanv, sca_w, sca_b, svec);
  k_final     <<<1152, 256, 0, stream>>>(attn_out, svec, proj_w, pout_w, dw_w,
                                         uh, zh, (float*)d_out);
}
